// T6_45784351375415
// MI455X (gfx1250) — matrix operand order repaired and checked by reading, not compiled, not run on hardware
//
#include <hip/hip_runtime.h>
#include <hip/hip_fp16.h>

typedef __attribute__((ext_vector_type(16))) _Float16 v16h;
typedef __attribute__((ext_vector_type(8)))  float    v8f;

#define B_  4
#define LC  256
#define LR  64
#define NQ  3
#define LQ  32
#define NP  3
#define LP  128
#define H_  256
#define V_  32000
#define M_  (B_*LR)     /* 256 rows of the GEMM / output */
#define K_  H_          /* 256 */
#define NB  64          /* output columns per GEMM block (4 N-tiles) */

// ---------------- kernel 1: feature max-pool over 7 sources, pack f16 A ----
__global__ __launch_bounds__(256) void k_fmax(
    const float* __restrict__ fc, const float* __restrict__ fq,
    const float* __restrict__ fp, _Float16* __restrict__ A)
{
  int i = blockIdx.x * 256 + threadIdx.x;     // over M_*K_ = 65536
  if (i >= M_ * K_) return;
  int h   = i % H_;
  int row = i / H_;                            // b*LR + r
  int b = row / LR, r = row % LR;
  float m = fc[i];
#pragma unroll
  for (int q = 0; q < NQ; ++q)
    m = fmaxf(m, fq[((b*NQ + q)*LR + r)*H_ + h]);
#pragma unroll
  for (int p = 0; p < NP; ++p)
    m = fmaxf(m, fp[((b*NP + p)*LR + r)*H_ + h]);
  A[i] = (_Float16)m;
}

// ---------------- kernel 2: logits = fmax @ W_gen + b_gen via WMMA f16 -----
// Block: 256 threads = 8 waves. Block covers full M (16 M-tiles) x NB cols
// (4 N-tiles). Wave w owns M-tiles {2w, 2w+1} x all 4 N-tiles.
// A panel (256 rows x 32 K) staged in LDS per K-step; W_gen streamed from
// HBM once, converted f32->f16 in registers into the B fragment layout.
__global__ __launch_bounds__(256) void k_gemm(
    const _Float16* __restrict__ A, const float* __restrict__ W,
    const float* __restrict__ bias, float* __restrict__ logits)
{
  __shared__ __align__(16) _Float16 As[M_][40];   // 32 K + 8 pad (80B stride)
  const int n0   = blockIdx.x * NB;
  const int tid  = threadIdx.x;
  const int lane = tid & 31, wv = tid >> 5;
  const int half = lane >> 4, l15 = lane & 15;

  v8f acc[2][4];
#pragma unroll
  for (int a = 0; a < 2; ++a)
#pragma unroll
    for (int c = 0; c < 4; ++c)
#pragma unroll
      for (int e = 0; e < 8; ++e) acc[a][c][e] = 0.f;

  for (int k0 = 0; k0 < K_; k0 += 32) {
    __syncthreads();
    {   // cooperative A panel load: thread t copies row t, 32 halves (64B)
      const uint4* s = (const uint4*)(A + (size_t)tid * K_ + k0);
      uint4* d = (uint4*)(&As[tid][0]);
      d[0] = s[0]; d[1] = s[1]; d[2] = s[2]; d[3] = s[3];
    }
    __syncthreads();

    // A fragments: lane(l15)=M row, half selects K chunks {h*8.., 16+h*8..}
    v16h af[2];
#pragma unroll
    for (int mt2 = 0; mt2 < 2; ++mt2) {
      const int m = (wv*2 + mt2)*16 + l15;
#pragma unroll
      for (int j = 0; j < 8; ++j) {
        af[mt2][j]     = As[m][half*8 + j];
        af[mt2][8 + j] = As[m][16 + half*8 + j];
      }
    }

    // B fragments: lane(l15)=N col, half selects K chunks {h*8.., 16+h*8..}; 16 V-strided loads
    // with fused f32->f16 conversion (W_gen read exactly once device-wide).
#pragma unroll
    for (int nt = 0; nt < 4; ++nt) {
      const int col = n0 + nt*16 + l15;
      const float* wp = W + (size_t)(k0 + half*8) * V_ + col;
      v16h bf;
#pragma unroll
      for (int j = 0; j < 16; ++j) bf[j] = (_Float16)wp[(size_t)((j & 7) + ((j >> 3) << 4)) * V_];
#pragma unroll
      for (int mt2 = 0; mt2 < 2; ++mt2)
        acc[mt2][nt] = __builtin_amdgcn_wmma_f32_16x16x32_f16(
            false, af[mt2], false, bf, (short)0, acc[mt2][nt], false, false);
    }
  }

  // C/D layout: VGPR r -> M = r + half*8, N = l15
#pragma unroll
  for (int nt = 0; nt < 4; ++nt) {
    const int col = n0 + nt*16 + l15;
    const float bv = bias[col];
#pragma unroll
    for (int mt2 = 0; mt2 < 2; ++mt2) {
      const int mbase = (wv*2 + mt2)*16 + half*8;
#pragma unroll
      for (int rr = 0; rr < 8; ++rr)
        logits[(size_t)(mbase + rr) * V_ + col] = acc[mt2][nt][rr] + bv;
    }
  }
}

// ---------------- kernel 3: per-row softmax stats (max, sum exp) -----------
__global__ __launch_bounds__(256) void k_rowstat(
    const float* __restrict__ logits, float* __restrict__ rmax,
    float* __restrict__ rsum)
{
  __shared__ float red[256];
  const int row = blockIdx.x, tid = threadIdx.x;
  const float* lp = logits + (size_t)row * V_;
  float m = -3.4e38f;
  for (int v = tid; v < V_; v += 256) m = fmaxf(m, lp[v]);
  red[tid] = m; __syncthreads();
  for (int s = 128; s > 0; s >>= 1) {
    if (tid < s) red[tid] = fmaxf(red[tid], red[tid + s]);
    __syncthreads();
  }
  m = red[0]; __syncthreads();
  float sum = 0.f;
  for (int v = tid; v < V_; v += 256) sum += __expf(lp[v] - m);
  red[tid] = sum; __syncthreads();
  for (int s = 128; s > 0; s >>= 1) {
    if (tid < s) red[tid] += red[tid + s];
    __syncthreads();
  }
  if (tid == 0) { rmax[row] = m; rsum[row] = red[0]; }
}

// ---------------- kernel 4: scatter-sum per source, max-pool via atomicMax -
// blockIdx.x = row (b*LR+r), blockIdx.y = source (0:ctx, 1..3:q, 4..6:p).
// Duplicate tokens summed by first-occurrence O(L) scan; max over sources is
// uint atomicMax (all values >= 0, d_out pre-zeroed).
__global__ __launch_bounds__(256) void k_scatter(
    const int* __restrict__ ctx, const int* __restrict__ qry,
    const int* __restrict__ psg, const float* __restrict__ cw,
    const float* __restrict__ qw, const float* __restrict__ pw,
    const float* __restrict__ selq, const float* __restrict__ selp,
    unsigned int* __restrict__ out)
{
  const int row = blockIdx.x;
  const int b = row / LR, r = row % LR;
  const int src = blockIdx.y;
  const int l = threadIdx.x;
  int L; const int* tok; const float* w; float gate;
  if (src == 0) {
    L = LC; tok = ctx + b*LC; w = cw + (size_t)row * LC; gate = 1.f;
  } else if (src <= NQ) {
    const int q = src - 1;
    L = LQ; tok = qry + (b*NQ + q)*LQ;
    w = qw + ((size_t)(b*NQ + q)*LR + r)*LQ;
    gate = selq[b*NQ + q];
  } else {
    const int p = src - 1 - NQ;
    L = LP; tok = psg + (b*NP + p)*LP;
    w = pw + ((size_t)(b*NP + p)*LR + r)*LP;
    gate = selp[b*NP + p];
  }
  if (l >= L) return;
  const int t = tok[l];
  float s = 0.f; bool first = true;
  for (int l2 = 0; l2 < L; ++l2) {
    if (tok[l2] == t) {
      if (l2 < l) { first = false; break; }
      s += w[l2];
    }
  }
  if (first)
    atomicMax(out + (size_t)row * V_ + t, __float_as_uint(gate * s));
}

// ---------------- kernel 5: out = 0.5*maxdist + 0.5*softmax ---------------
__global__ __launch_bounds__(256) void k_final(
    const float* __restrict__ logits, const float* __restrict__ rmax,
    const float* __restrict__ rsum, float* __restrict__ out)
{
  const size_t i4 = ((size_t)blockIdx.x * 256 + threadIdx.x) * 4;
  if (i4 >= (size_t)M_ * V_) return;
  const int row = (int)(i4 / V_);               // V_ % 4 == 0: same row for all 4
  const float m = rmax[row], rinv = 0.5f / rsum[row];
  float4 lg = *(const float4*)(logits + i4);
  float4 o  = *(const float4*)(out + i4);       // bits == positive floats
  o.x = 0.5f*o.x + __expf(lg.x - m)*rinv;
  o.y = 0.5f*o.y + __expf(lg.y - m)*rinv;
  o.z = 0.5f*o.z + __expf(lg.z - m)*rinv;
  o.w = 0.5f*o.w + __expf(lg.w - m)*rinv;
  *(float4*)(out + i4) = o;
}

extern "C" void kernel_launch(void* const* d_in, const int* in_sizes, int n_in,
                              void* d_out, int out_size, void* d_ws, size_t ws_size,
                              hipStream_t stream)
{
  (void)in_sizes; (void)n_in; (void)out_size; (void)ws_size;
  const int*   ctx  = (const int*)  d_in[0];
  const int*   qry  = (const int*)  d_in[1];
  const int*   psg  = (const int*)  d_in[2];
  const float* cw   = (const float*)d_in[3];
  const float* qw   = (const float*)d_in[4];
  const float* pw   = (const float*)d_in[5];
  const float* fc   = (const float*)d_in[6];
  const float* fq   = (const float*)d_in[7];
  const float* fp   = (const float*)d_in[8];
  const float* selq = (const float*)d_in[9];
  const float* selp = (const float*)d_in[10];
  const float* Wg   = (const float*)d_in[11];
  const float* bg   = (const float*)d_in[12];
  float* out = (float*)d_out;

  char* ws = (char*)d_ws;
  _Float16* Ah  = (_Float16*)ws;                               // 128 KB
  float* logits = (float*)(ws + (256 << 10));                  // 32.77 MB
  float* rmax   = (float*)(ws + (256 << 10) + (size_t)M_ * V_ * sizeof(float));
  float* rsum   = rmax + M_;

  hipMemsetAsync(d_out, 0, (size_t)M_ * V_ * sizeof(float), stream);

  k_fmax<<<(M_ * K_) / 256, 256, 0, stream>>>(fc, fq, fp, Ah);
  k_gemm<<<V_ / NB, 256, 0, stream>>>(Ah, Wg, bg, logits);
  k_rowstat<<<M_, 256, 0, stream>>>(logits, rmax, rsum);
  dim3 sg(M_, 1 + NQ + NP);
  k_scatter<<<sg, 256, 0, stream>>>(ctx, qry, psg, cw, qw, pw, selq, selp,
                                    (unsigned int*)d_out);
  k_final<<<(M_ * (V_ / 4)) / 256, 256, 0, stream>>>(logits, rmax, rsum, out);
}
